// ZSequenceGenerator_31104153157940
// MI455X (gfx1250) — compile-verified, matrix operand order checked by reading, not run on hardware
//
#include <hip/hip_runtime.h>
#include <hip/hip_bf16.h>
#include <stdint.h>

typedef __attribute__((ext_vector_type(16))) __bf16 v16bf;
typedef __attribute__((ext_vector_type(8)))  float  v8f;
typedef __attribute__((ext_vector_type(4)))  unsigned int u32x4;
typedef __attribute__((ext_vector_type(8)))  int i32x8;
typedef __attribute__((ext_vector_type(4)))  int i32x4;

#define BATCH   16384
#define INPUT_D 1024
#define HID_D   1024
#define LAT_D   256
#define KSEG    8

#define HAS_TDM __has_builtin(__builtin_amdgcn_tensor_load_to_lds)

union FragBF { v16bf v; uint4 u[2]; };

#if HAS_TDM
// ---------------------------------------------------------------------------
// TDM 2D tile load: 128 rows x 32 bf16 from global (row pitch lda elements)
// into LDS at lds_off with +16B pad per 64B row (pitch 80B = 40 bf16).
// D# packing per CDNA5 ISA §8.3/§8.4 (group0 128b, group1 256b, 2D tile).
// amdgpu-toolchain (clang-23) builtin form: 6 args, extra i32x8 zero-filled.
// ---------------------------------------------------------------------------
__device__ __forceinline__ void tdm_tile_load(unsigned lds_off,
                                              const __bf16* gsrc, int lda)
{
    unsigned long long ga = (unsigned long long)(uintptr_t)gsrc;
    u32x4 g0;
    g0[0] = 1u;                                   // count=1 valid, user desc
    g0[1] = lds_off;                              // lds_addr (bytes)
    g0[2] = (unsigned)ga;                         // global_addr[31:0]
    g0[3] = (unsigned)((ga >> 32) & 0x01ffffffu)  // global_addr[56:32]
          | 0x80000000u;                          // type = 2 ("image")
    i32x8 g1;
    g1[0] = (int)((1u << 16)                      // data_size = 2 bytes
                | (1u << 20)                      // pad_enable
                | (3u << 22)                      // pad_interval: every 16 DW
                | (3u << 25));                    // pad_amount: 4 DW (16B)
    g1[1] = (int)(32u << 16);                     // tensor_dim0 = 32 (lo16)
    g1[2] = (int)(128u << 16);                    // td0 hi | tensor_dim1 = 128
    g1[3] = (int)(32u << 16);                     // td1 hi | tile_dim0 = 32
    g1[4] = (int)128u;                            // tile_dim1 = 128, tile_dim2=0
    g1[5] = (int)(unsigned)lda;                   // tensor_dim0_stride[31:0]
    g1[6] = 0;                                    // stride hi | dim1_stride lo
    g1[7] = 0;                                    // dim1_stride hi
    i32x4 gz4 = {0, 0, 0, 0};
    i32x8 gz8 = {0, 0, 0, 0, 0, 0, 0, 0};
    __builtin_amdgcn_tensor_load_to_lds(g0, g1, gz4, gz4, gz8, 0);
}
#endif

// 8 WMMAs on one 32-wide K slab held in LDS (padded pitch 40 bf16)
__device__ __forceinline__ void wmma_step(const __bf16 (*sa)[40],
                                          const __bf16 (*sb)[40],
                                          v8f acc[4][2],
                                          int wm, int wn, int frow, int kb)
{
    FragBF af[4], bfr[2];
    #pragma unroll
    for (int i = 0; i < 4; ++i) {
        const __bf16* p = &sa[wm + 16*i + frow][kb];
        af[i].u[0] = *reinterpret_cast<const uint4*>(p);
        af[i].u[1] = *reinterpret_cast<const uint4*>(p + 16);
    }
    #pragma unroll
    for (int j = 0; j < 2; ++j) {
        const __bf16* p = &sb[wn + 16*j + frow][kb];
        bfr[j].u[0] = *reinterpret_cast<const uint4*>(p);
        bfr[j].u[1] = *reinterpret_cast<const uint4*>(p + 16);
    }
    #pragma unroll
    for (int i = 0; i < 4; ++i)
        #pragma unroll
        for (int j = 0; j < 2; ++j)
            acc[i][j] = __builtin_amdgcn_wmma_f32_16x16x32_bf16(
                false, af[i].v, false, bfr[j].v, (short)0, acc[i][j],
                false, false);
}

// ---------------------------------------------------------------------------
// bf16 GEMM: C = act(A @ W^T + bias). A: M x Kd (row stride lda), W: N x Kd.
// 128x128 block tile, K-step 32, double-buffered LDS filled by TDM
// (wave 0 issues tensor_load_to_lds; s_wait_tensorcnt + barrier publishes).
// Optional f32 and/or bf16 outputs.
// ---------------------------------------------------------------------------
template <int ACT, int WF32, int WBF16>
__global__ __launch_bounds__(256)
void gemm_bf16(const __bf16* __restrict__ A, int lda,
               const __bf16* __restrict__ W,
               const float*  __restrict__ bias,
               float*  __restrict__ Cf, int ldcf,
               __bf16* __restrict__ Cb, int ldcb,
               int M, int N, int Kd)
{
    __shared__ __align__(16) __bf16 sA[2][128][40];
    __shared__ __align__(16) __bf16 sB[2][128][40];

    const int bm   = blockIdx.y * 128;
    const int bn   = blockIdx.x * 128;
    const int tid  = threadIdx.x;
    const int wave = tid >> 5;
    const int lane = tid & 31;
    const int wm   = (wave >> 2) * 64;
    const int wn   = (wave & 3) * 32;
    const int frow = lane & 15;
    const int kb   = (lane >> 4) * 8;

    v8f acc[4][2] = {};
    const int nk = Kd >> 5;

#if HAS_TDM
    const unsigned ldsA0 = (unsigned)(uintptr_t)&sA[0][0][0];
    const unsigned ldsA1 = (unsigned)(uintptr_t)&sA[1][0][0];
    const unsigned ldsB0 = (unsigned)(uintptr_t)&sB[0][0][0];
    const unsigned ldsB1 = (unsigned)(uintptr_t)&sB[1][0][0];
    const __bf16* Ab = A + (size_t)bm * lda;
    const __bf16* Wb = W + (size_t)bn * Kd;

    if (wave == 0) {                       // prologue: fill buffer 0
        tdm_tile_load(ldsA0, Ab, lda);
        tdm_tile_load(ldsB0, Wb, Kd);
    }
    for (int t = 0; t < nk; ++t) {
        const int cur = t & 1;
        if (wave == 0) {
            if (t + 1 < nk) {              // prefetch next slab into other buf
                const int kn = (t + 1) << 5;
                tdm_tile_load(cur ? ldsA0 : ldsA1, Ab + kn, lda);
                tdm_tile_load(cur ? ldsB0 : ldsB1, Wb + kn, Kd);
                __builtin_amdgcn_s_wait_tensorcnt(2);   // cur pair landed
            } else {
                __builtin_amdgcn_s_wait_tensorcnt(0);
            }
        }
        __syncthreads();                   // publish cur slab to all waves
        wmma_step(sA[cur], sB[cur], acc, wm, wn, frow, kb);
        __syncthreads();                   // all reads done before overwrite
    }
#else
    // fallback: cooperative bf16 global->LDS staging (single buffer)
    const int r     = tid >> 1;
    const int halfk = (tid & 1) * 16;
    for (int t = 0; t < nk; ++t) {
        const int k0 = t << 5;
        const uint4* ap = reinterpret_cast<const uint4*>(
            A + (size_t)(bm + r) * lda + k0 + halfk);
        const uint4* wp = reinterpret_cast<const uint4*>(
            W + (size_t)(bn + r) * Kd + k0 + halfk);
        uint4 a0 = ap[0], a1 = ap[1], w0 = wp[0], w1 = wp[1];
        __syncthreads();
        *reinterpret_cast<uint4*>(&sA[0][r][halfk])     = a0;
        *reinterpret_cast<uint4*>(&sA[0][r][halfk + 8]) = a1;
        *reinterpret_cast<uint4*>(&sB[0][r][halfk])     = w0;
        *reinterpret_cast<uint4*>(&sB[0][r][halfk + 8]) = w1;
        __syncthreads();
        wmma_step(sA[0], sB[0], acc, wm, wn, frow, kb);
    }
#endif

    // epilogue: C/D layout — VGPR v: lanes 0-15 -> M=v, lanes 16-31 -> M=v+8
    const int mh   = (lane < 16) ? 0 : 8;
    const int ccol = lane & 15;
    #pragma unroll
    for (int i = 0; i < 4; ++i) {
        #pragma unroll
        for (int j = 0; j < 2; ++j) {
            #pragma unroll
            for (int vv = 0; vv < 8; ++vv) {
                const int m = bm + wm + 16*i + vv + mh;
                const int n = bn + wn + 16*j + ccol;
                float val = acc[i][j][vv] + bias[n];
                if (ACT == 1)
                    val = 0.5f * val * (1.0f + erff(val * 0.70710678118654752f));
                if (WF32)  Cf[(size_t)m * ldcf + n] = val;
                if (WBF16) Cb[(size_t)m * ldcb + n] = (__bf16)val;
            }
        }
    }
}

// ---------------------------------------------------------------------------
__global__ __launch_bounds__(256)
void cvt_f32_bf16(const float* __restrict__ s, __bf16* __restrict__ d, int n)
{
    int i = blockIdx.x * 256 + threadIdx.x;
    if (i < n) d[i] = (__bf16)s[i];
}

// gates ordered (r,u,n): r,u use gi+gh summed; n = tanh(in + r*hn)
__global__ __launch_bounds__(256)
void gru_update(const float* __restrict__ Gi, const float* __restrict__ Gh,
                const float* __restrict__ h, float* __restrict__ hnew,
                __bf16* __restrict__ hnewb, int total)
{
    int idx = blockIdx.x * 256 + threadIdx.x;
    if (idx >= total) return;
    int b = idx >> 10;
    int n = idx & (HID_D - 1);
    size_t base = (size_t)b * (3 * HID_D);
    float ir = Gi[base + n],             hr = Gh[base + n];
    float iu = Gi[base + HID_D + n],     hu = Gh[base + HID_D + n];
    float in_ = Gi[base + 2*HID_D + n],  hn = Gh[base + 2*HID_D + n];
    float rg = 1.0f / (1.0f + __expf(-(ir + hr)));
    float ug = 1.0f / (1.0f + __expf(-(iu + hu)));
    float ng = tanhf(in_ + rg * hn);
    float hv = (1.0f - ug) * ng + ug * h[idx];
    hnew[idx]  = hv;
    hnewb[idx] = (__bf16)hv;
}

__global__ __launch_bounds__(256)
void halt_step(const float* __restrict__ hnew, const float* __restrict__ Wh,
               const float* __restrict__ bh,
               float* __restrict__ cum, float* __restrict__ nseg,
               float* __restrict__ halts, float* __restrict__ weights,
               float* __restrict__ numseg_out, int k)
{
    const int wave = threadIdx.x >> 5;
    const int lane = threadIdx.x & 31;
    const int b = blockIdx.x * 8 + wave;
    if (b >= BATCH) return;
    const float* hp = hnew + (size_t)b * HID_D;
    float s = 0.f;
    #pragma unroll
    for (int j = 0; j < HID_D / 32; ++j) s += hp[lane + 32*j] * Wh[lane + 32*j];
    #pragma unroll
    for (int off = 16; off; off >>= 1) s += __shfl_xor(s, off, 32);
    if (lane == 0) {
        float lam = 1.0f / (1.0f + __expf(-(s + bh[0])));
        float c   = cum[b];
        float nc  = c + lam;
        float rem = 1.0f - c;
        float w   = (c >= 1.0f) ? 0.0f
                  : ((nc >= 1.0f) ? fmaxf(rem, 0.01f) : 1.0f);
        halts[(size_t)b * KSEG + k]   = lam;
        weights[(size_t)b * KSEG + k] = w;
        cum[b] = nc;
        float ns = nseg[b] + w;
        nseg[b] = ns;
        numseg_out[b] = ns;
    }
}

__global__ void halt_init(float* cum, float* nseg, float* halts,
                          float* weights, float* numseg_out)
{
    int b = blockIdx.x * 256 + threadIdx.x;
    if (b >= BATCH) return;
    cum[b] = 0.0f;
    nseg[b] = 1.0f;
    halts[(size_t)b * KSEG]   = 0.0f;
    weights[(size_t)b * KSEG] = 1.0f;
    numseg_out[b] = 1.0f;
}

// ---------------------------------------------------------------------------
extern "C" void kernel_launch(void* const* d_in, const int* in_sizes, int n_in,
                              void* d_out, int out_size, void* d_ws, size_t ws_size,
                              hipStream_t stream)
{
    (void)in_sizes; (void)n_in; (void)out_size; (void)ws_size;
    const float* emb    = (const float*)d_in[0];
    const float* W_in1  = (const float*)d_in[1];
    const float* b_in1  = (const float*)d_in[2];
    const float* W_in2  = (const float*)d_in[3];
    const float* b_in2  = (const float*)d_in[4];
    const float* W_ih   = (const float*)d_in[5];
    const float* b_ih   = (const float*)d_in[6];
    const float* W_hh   = (const float*)d_in[7];
    const float* b_hh   = (const float*)d_in[8];
    const float* W_z    = (const float*)d_in[9];
    const float* b_z    = (const float*)d_in[10];
    const float* W_halt = (const float*)d_in[11];
    const float* b_halt = (const float*)d_in[12];

    float* out     = (float*)d_out;
    float* zseq    = out;                                   // B*K*LAT
    float* halts   = zseq + (size_t)BATCH * KSEG * LAT_D;   // B*K
    float* weights = halts + (size_t)BATCH * KSEG;          // B*K
    float* numseg  = weights + (size_t)BATCH * KSEG;        // B

    // f32 workspace
    float* hAf  = (float*)d_ws;                         // B*HID
    float* hBf  = hAf + (size_t)BATCH * HID_D;          // B*HID
    float* Gi   = hBf + (size_t)BATCH * HID_D;          // B*3H
    float* Gh   = Gi + (size_t)BATCH * 3 * HID_D;       // B*3H
    float* cum  = Gh + (size_t)BATCH * 3 * HID_D;       // B
    float* nseg = cum + BATCH;                          // B
    // bf16 workspace
    __bf16* embb  = (__bf16*)(nseg + BATCH);
    __bf16* h1b   = embb + (size_t)BATCH * INPUT_D;     // B*HID
    __bf16* hAb   = h1b  + (size_t)BATCH * HID_D;       // B*HID
    __bf16* hBb   = hAb  + (size_t)BATCH * HID_D;       // B*HID
    __bf16* zb    = hBb  + (size_t)BATCH * HID_D;       // B*LAT
    __bf16* w1b   = zb   + (size_t)BATCH * LAT_D;       // HID*INPUT
    __bf16* w2b   = w1b  + (size_t)HID_D * INPUT_D;     // HID*HID
    __bf16* wihb  = w2b  + (size_t)HID_D * HID_D;       // 3H*LAT
    __bf16* whhb  = wihb + (size_t)3 * HID_D * LAT_D;   // 3H*HID
    __bf16* wzb   = whhb + (size_t)3 * HID_D * HID_D;   // LAT*HID

    const dim3 blk(256);
    const dim3 gHID(HID_D / 128, BATCH / 128);
    const dim3 gGAT(3 * HID_D / 128, BATCH / 128);
    const dim3 gLAT(LAT_D / 128, BATCH / 128);
    auto cgrid = [](size_t n) { return dim3((unsigned)((n + 255) / 256)); };

    // one-time (per launch) bf16 conversions: input + all weights
    cvt_f32_bf16<<<cgrid((size_t)BATCH * INPUT_D), blk, 0, stream>>>(
        emb, embb, BATCH * INPUT_D);
    cvt_f32_bf16<<<cgrid((size_t)HID_D * INPUT_D), blk, 0, stream>>>(
        W_in1, w1b, HID_D * INPUT_D);
    cvt_f32_bf16<<<cgrid((size_t)HID_D * HID_D), blk, 0, stream>>>(
        W_in2, w2b, HID_D * HID_D);
    cvt_f32_bf16<<<cgrid((size_t)3 * HID_D * LAT_D), blk, 0, stream>>>(
        W_ih, wihb, 3 * HID_D * LAT_D);
    cvt_f32_bf16<<<cgrid((size_t)3 * HID_D * HID_D), blk, 0, stream>>>(
        W_hh, whhb, 3 * HID_D * HID_D);
    cvt_f32_bf16<<<cgrid((size_t)LAT_D * HID_D), blk, 0, stream>>>(
        W_z, wzb, LAT_D * HID_D);

    // input_proj: H1 = gelu(E @ W_in1^T + b_in1)   (bf16-only output)
    gemm_bf16<1, 0, 1><<<gHID, blk, 0, stream>>>(
        embb, INPUT_D, w1b, b_in1, nullptr, 0, h1b, HID_D,
        BATCH, HID_D, INPUT_D);
    // H = H1 @ W_in2^T + b_in2                     (f32 + bf16 outputs)
    gemm_bf16<0, 1, 1><<<gHID, blk, 0, stream>>>(
        h1b, HID_D, w2b, b_in2, hAf, HID_D, hAb, HID_D,
        BATCH, HID_D, HID_D);
    // z0 = H @ W_z^T + b_z -> z_seq[:,0,:] (f32, strided) + compact bf16 copy
    gemm_bf16<0, 1, 1><<<gLAT, blk, 0, stream>>>(
        hAb, HID_D, wzb, b_z, zseq, KSEG * LAT_D, zb, LAT_D,
        BATCH, LAT_D, HID_D);
    halt_init<<<cgrid(BATCH), blk, 0, stream>>>(cum, nseg, halts, weights, numseg);

    float*  hcurf = hAf; float*  hnewf = hBf;
    __bf16* hcurb = hAb; __bf16* hnewb = hBb;
    for (int k = 1; k < KSEG; ++k) {
        // Gi = z @ W_ih^T + b_ih
        gemm_bf16<0, 1, 0><<<gGAT, blk, 0, stream>>>(
            zb, LAT_D, wihb, b_ih, Gi, 3 * HID_D, nullptr, 0,
            BATCH, 3 * HID_D, LAT_D);
        // Gh = h @ W_hh^T + b_hh
        gemm_bf16<0, 1, 0><<<gGAT, blk, 0, stream>>>(
            hcurb, HID_D, whhb, b_hh, Gh, 3 * HID_D, nullptr, 0,
            BATCH, 3 * HID_D, HID_D);
        gru_update<<<cgrid((size_t)BATCH * HID_D), blk, 0, stream>>>(
            Gi, Gh, hcurf, hnewf, hnewb, BATCH * HID_D);
        // z_k = h_new @ W_z^T + b_z
        gemm_bf16<0, 1, 1><<<gLAT, blk, 0, stream>>>(
            hnewb, HID_D, wzb, b_z, zseq + (size_t)k * LAT_D, KSEG * LAT_D,
            zb, LAT_D, BATCH, LAT_D, HID_D);
        halt_step<<<dim3(BATCH / 8), blk, 0, stream>>>(
            hnewf, W_halt, b_halt, cum, nseg, halts, weights, numseg, k);
        float* tf = hcurf; hcurf = hnewf; hnewf = tf;
        __bf16* tb = hcurb; hcurb = hnewb; hnewb = tb;
    }
}
